// HighTopicAttention_80805514707579
// MI455X (gfx1250) — matrix operand order repaired and checked by reading, not compiled, not run on hardware
//
#include <hip/hip_runtime.h>
#include <cstddef>

// Problem constants (match reference)
#define BB 64
#define TT 512
#define DD 1024
#define NTOP 32
#define EPSV 1e-7f

typedef __attribute__((ext_vector_type(16))) __bf16 v16bf;
typedef __attribute__((ext_vector_type(8)))  __bf16 v8bf;
typedef __attribute__((ext_vector_type(8)))  float  v8f;

// ---------------- helpers ----------------

__device__ inline __bf16 f2bf(float f) {
  union { float f; unsigned int u; } in; in.f = f;
  unsigned int u = in.u;
  u += 0x7FFFu + ((u >> 16) & 1u);          // round-to-nearest-even
  union { unsigned short s; __bf16 b; } out; out.s = (unsigned short)(u >> 16);
  return out.b;
}

__device__ inline v8f wmma_bf16(v16bf a, v16bf b, v8f c) {
  // (neg_a, A, neg_b, B, c_mod, C, reuse_a, reuse_b)
  return __builtin_amdgcn_wmma_f32_16x16x32_bf16(false, a, false, b, (short)0, c,
                                                 false, false);
}

// Generic 16-element bf16 fragment gather: two 16B chunks at p and p+hi_off.
// A (row-major, stride DD): lane base = row*DD + ((lane>=16)?16:0), hi_off = 8.
//   -> lanes 0-15: M=lane, K 0..15; lanes 16-31: K 16..31
// B (column-major, stride DD): lane base = col*DD + ((lane>=16)?16:0), hi_off = 8.
//   -> lanes 0-15: N=lane, K 0..15; lanes 16-31: K 16..31
__device__ inline v16bf frag16(const __bf16* p, int hi_off) {
  v8bf lo = *(const v8bf*)(p);
  v8bf hi = *(const v8bf*)(p + hi_off);
  v16bf f;
#pragma unroll
  for (int i = 0; i < 8; ++i) { f[i] = lo[i]; f[i + 8] = hi[i]; }
  return f;
}

// ---------------- K0a: one-time weight transpose + bf16 convert ----------------
// out[n*1024 + k] = bf16(in[k*1024 + n])  (runs once per weight; L2-resident after)
__global__ __launch_bounds__(256) void wprep_kernel(const float* __restrict__ in,
                                                    __bf16* __restrict__ out) {
  __shared__ float tile[32][33];
  const int k0 = blockIdx.x * 32, n0 = blockIdx.y * 32;
  const int tx = threadIdx.x & 31, ty = threadIdx.x >> 5;   // ty 0..7
#pragma unroll
  for (int r = 0; r < 4; ++r)
    tile[ty + r * 8][tx] = in[(k0 + ty + r * 8) * DD + n0 + tx];
  __syncthreads();
#pragma unroll
  for (int r = 0; r < 4; ++r)
    out[(size_t)(n0 + ty + r * 8) * DD + k0 + tx] = f2bf(tile[tx][ty + r * 8]);
}

// ---------------- K0b: plain f32 -> bf16 copy (row-major A operands) ----------------
__global__ __launch_bounds__(256) void abf_kernel(const float* __restrict__ in,
                                                  __bf16* __restrict__ out, int n) {
  int i = blockIdx.x * 256 + threadIdx.x;
  if (i < n) out[i] = f2bf(in[i]);
}

// ---------------- K1/K2: M=64 GEMM (+bias, +tanh) via WMMA ----------------
// Out[64 x 1024] = act(Abf[64 x 1024] @ B + bias); Abf bf16 row-major,
// BT bf16 pre-transposed [N x K]. No LDS/barriers; A+B both ping-pong pipelined.
#define GK_TN 128

__global__ __launch_bounds__(256) void gemm64_kernel(
    const __bf16* __restrict__ Abf, const __bf16* __restrict__ BT,
    const float* __restrict__ bias, float* __restrict__ Out,
    __bf16* __restrict__ OutBf, int act) {
  const int tid  = threadIdx.x;
  const int wave = tid >> 5;
  const int lane = tid & 31;
  const int n0   = blockIdx.x * GK_TN;
  const int rt   = wave & 3;                 // row tile 0..3 (16 rows each)
  const int cg   = wave >> 2;                // col group 0/1 -> 4 col tiles each
  const __bf16* abase = Abf + (size_t)(rt * 16 + (lane & 15)) * DD + ((lane >> 4) << 4);
  const __bf16* bbase = BT  + (size_t)(n0 + cg * 64 + (lane & 15)) * DD + ((lane >> 4) << 4);

  v8f acc[4] = {v8f{}, v8f{}, v8f{}, v8f{}};
  v16bf bfA[4], bfB[4], afA, afB;
  afA = frag16(abase, 8);
#pragma unroll
  for (int j = 0; j < 4; ++j) bfA[j] = frag16(bbase + (size_t)j * 16 * DD, 8);

  // ping-pong pipeline over k in steps of 64 (2 WMMA k-steps)
  int k0 = 0;
  for (; k0 < DD - 64; k0 += 64) {
    afB = frag16(abase + k0 + 32, 8);
#pragma unroll
    for (int j = 0; j < 4; ++j)
      bfB[j] = frag16(bbase + (size_t)j * 16 * DD + k0 + 32, 8);
#pragma unroll
    for (int j = 0; j < 4; ++j) acc[j] = wmma_bf16(afA, bfA[j], acc[j]);

    afA = frag16(abase + k0 + 64, 8);
#pragma unroll
    for (int j = 0; j < 4; ++j)
      bfA[j] = frag16(bbase + (size_t)j * 16 * DD + k0 + 64, 8);
#pragma unroll
    for (int j = 0; j < 4; ++j) acc[j] = wmma_bf16(afB, bfB[j], acc[j]);
  }
  // tail: k0 == DD-64 -> two k-steps left
  afB = frag16(abase + k0 + 32, 8);
#pragma unroll
  for (int j = 0; j < 4; ++j)
    bfB[j] = frag16(bbase + (size_t)j * 16 * DD + k0 + 32, 8);
#pragma unroll
  for (int j = 0; j < 4; ++j) acc[j] = wmma_bf16(afA, bfA[j], acc[j]);
#pragma unroll
  for (int j = 0; j < 4; ++j) acc[j] = wmma_bf16(afB, bfB[j], acc[j]);

  // epilogue: C/D layout -> row = rt*16 + ((lane>=16)?8:0) + v, col = lane&15
  const int rofs = (lane >> 4) << 3;
  const int cin  = lane & 15;
#pragma unroll
  for (int j = 0; j < 4; ++j) {
    int col = n0 + (cg * 4 + j) * 16 + cin;
    float bv = bias ? bias[col] : 0.0f;
#pragma unroll
    for (int v = 0; v < 8; ++v) {
      float val = acc[j][v] + bv;
      if (act) val = tanhf(val);
      int idx = (rt * 16 + rofs + v) * DD + col;
      Out[idx] = val;
      if (OutBf) OutBf[idx] = f2bf(val);
    }
  }
}

// ---------------- K3: topic softmax + uw ----------------
__global__ __launch_bounds__(256) void topic_kernel(
    const float* __restrict__ tij, const float* __restrict__ topics,
    float* __restrict__ uw) {
  __shared__ float part[NTOP][8];
  __shared__ float at[NTOP];
  __shared__ float inv_s;
  const int b = blockIdx.x, tid = threadIdx.x;
  const int k = tid >> 3, s = tid & 7;
  const float* trow = topics + k * DD;
  const float* xrow = tij + b * DD;
  float p = 0.f;
  for (int i = s; i < DD; i += 8) p += xrow[i] * trow[i];
  part[k][s] = p;
  __syncthreads();
  if (tid < NTOP) {
    float ts = 0.f;
#pragma unroll
    for (int i = 0; i < 8; ++i) ts += part[tid][i];
    at[tid] = ts;
  }
  __syncthreads();
  if (tid == 0) {
    float m = at[0];
    for (int i = 1; i < NTOP; ++i) m = fmaxf(m, at[i]);
    float sum = 0.f;
    for (int i = 0; i < NTOP; ++i) { float e = __expf(at[i] - m); at[i] = e; sum += e; }
    inv_s = 1.0f / (sum + EPSV);
  }
  __syncthreads();
  float inv = inv_s;
#pragma unroll
  for (int i = 0; i < 4; ++i) {
    int d = tid + i * 256;
    float acc = 0.f;
#pragma unroll
    for (int kk = 0; kk < NTOP; ++kk) acc += at[kk] * topics[kk * DD + d];
    uw[b * DD + d] = acc * inv;
  }
}

// ---------------- K4: fused logits e[b,t] = sum_d tanh(xW + uitc + b)*uw ----------------
// x tile resident in LDS (bf16); W-fragments streamed global->VGPR from the
// pre-transposed bf16 copy (L2-resident). A+B ping-pong pipeline, no barriers
// in main loop, 8 independent WMMA chains per wave.
#define XC_TM 16
#define XSTRIDE (DD + 8)   // 1032 elements

__global__ __launch_bounds__(256) void attn_logits_kernel(
    const float* __restrict__ x, const __bf16* __restrict__ WT,
    const float* __restrict__ bvec, const float* __restrict__ uitc,
    const float* __restrict__ uw, float* __restrict__ e) {
  __shared__ __bf16 sX[XC_TM * XSTRIDE];            // 33 KB, resident all kernel
  __shared__ float eacc[XC_TM];
  const int tid  = threadIdx.x;
  const int wave = tid >> 5;
  const int b    = blockIdx.y;
  const int t0   = blockIdx.x * XC_TM;
  if (tid < XC_TM) eacc[tid] = 0.f;
  {  // x tile: 16 rows x 1024 cols (f32 -> bf16), coalesced
    const float* xr = x + ((size_t)b * TT + t0) * DD;
#pragma unroll 8
    for (int j = 0; j < 64; ++j) {
      int lin = tid + j * 256;
      int row = lin >> 10, col = lin & 1023;
      sX[row * XSTRIDE + col] = f2bf(xr[row * DD + col]);
    }
  }
  __syncthreads();
  const int lane = tid & 31;
  const __bf16* xbase = sX + (lane & 15) * XSTRIDE + ((lane >> 4) << 4);
  const __bf16* wbase = WT + (size_t)(wave * 16 + (lane & 15)) * DD + ((lane >> 4) << 4);

  // wave w owns n-tiles {w, w+8, ..., w+56}: col base = j*128 + w*16
  v8f acc[8] = {v8f{}, v8f{}, v8f{}, v8f{}, v8f{}, v8f{}, v8f{}, v8f{}};
  v16bf bfA[8], bfB[8], afA, afB;
  afA = frag16(xbase, 8);
#pragma unroll
  for (int j = 0; j < 8; ++j) bfA[j] = frag16(wbase + (size_t)j * 128 * DD, 8);

  // ping-pong pipeline over k in steps of 64 (2 WMMA k-steps)
  int k0 = 0;
  for (; k0 < DD - 64; k0 += 64) {
    afB = frag16(xbase + k0 + 32, 8);
#pragma unroll
    for (int j = 0; j < 8; ++j)
      bfB[j] = frag16(wbase + (size_t)j * 128 * DD + k0 + 32, 8);
#pragma unroll
    for (int j = 0; j < 8; ++j) acc[j] = wmma_bf16(afA, bfA[j], acc[j]);

    afA = frag16(xbase + k0 + 64, 8);
#pragma unroll
    for (int j = 0; j < 8; ++j)
      bfA[j] = frag16(wbase + (size_t)j * 128 * DD + k0 + 64, 8);
#pragma unroll
    for (int j = 0; j < 8; ++j) acc[j] = wmma_bf16(afB, bfB[j], acc[j]);
  }
  // tail: k0 == DD-64 -> two k-steps left
  afB = frag16(xbase + k0 + 32, 8);
#pragma unroll
  for (int j = 0; j < 8; ++j)
    bfB[j] = frag16(wbase + (size_t)j * 128 * DD + k0 + 32, 8);
#pragma unroll
  for (int j = 0; j < 8; ++j) acc[j] = wmma_bf16(afA, bfA[j], acc[j]);
#pragma unroll
  for (int j = 0; j < 8; ++j) acc[j] = wmma_bf16(afB, bfB[j], acc[j]);

  // epilogue: tanh(S + uitc + b) * uw, reduce over d into eacc[row]
  const int rofs = (lane >> 4) << 3;
  const int cin  = lane & 15;
#pragma unroll
  for (int j = 0; j < 8; ++j) {
    int d = j * 128 + wave * 16 + cin;
    float uc = uitc[b * DD + d] + bvec[d];
    float w8 = uw[b * DD + d];
#pragma unroll
    for (int v = 0; v < 8; ++v) {
      float h = tanhf(acc[j][v] + uc);
      atomicAdd(&eacc[rofs + v], h * w8);
    }
  }
  __syncthreads();
  if (tid < XC_TM) e[(size_t)b * TT + t0 + tid] = eacc[tid];
}

// ---------------- K5: sequence softmax + out = a @ x ----------------
__global__ __launch_bounds__(256) void attn_out_kernel(
    const float* __restrict__ x, const float* __restrict__ e,
    float* __restrict__ out) {
  __shared__ float a[TT];
  __shared__ float red[256];
  const int b = blockIdx.x, tid = threadIdx.x;
  float v0 = e[b * TT + tid];
  float v1 = e[b * TT + 256 + tid];
  red[tid] = fmaxf(v0, v1);
  __syncthreads();
  for (int s = 128; s > 0; s >>= 1) {
    if (tid < s) red[tid] = fmaxf(red[tid], red[tid + s]);
    __syncthreads();
  }
  float m = red[0];
  __syncthreads();
  float e0 = __expf(v0 - m), e1 = __expf(v1 - m);
  red[tid] = e0 + e1;
  __syncthreads();
  for (int s = 128; s > 0; s >>= 1) {
    if (tid < s) red[tid] += red[tid + s];
    __syncthreads();
  }
  float inv = 1.0f / (red[0] + EPSV);
  a[tid] = e0 * inv;
  a[tid + 256] = e1 * inv;
  __syncthreads();
  float a0 = 0.f, a1 = 0.f, a2 = 0.f, a3 = 0.f;
  const float* xb = x + (size_t)b * TT * DD;
  for (int t = 0; t < TT; ++t) {
    const float* xr = xb + (size_t)t * DD;
    if (t + 8 < TT) __builtin_prefetch(xb + (size_t)(t + 8) * DD + tid, 0, 1);
    float av = a[t];
    a0 += av * xr[tid];
    a1 += av * xr[tid + 256];
    a2 += av * xr[tid + 512];
    a3 += av * xr[tid + 768];
  }
  out[b * DD + tid]       = a0;
  out[b * DD + tid + 256] = a1;
  out[b * DD + tid + 512] = a2;
  out[b * DD + tid + 768] = a3;
}

// ---------------- launcher ----------------
extern "C" void kernel_launch(void* const* d_in, const int* in_sizes, int n_in,
                              void* d_out, int out_size, void* d_ws, size_t ws_size,
                              hipStream_t stream) {
  const float* x      = (const float*)d_in[0];  // [64,512,1024]
  const float* c      = (const float*)d_in[1];  // [64,1024]
  const float* W      = (const float*)d_in[2];  // [1024,1024]
  const float* Wc     = (const float*)d_in[3];  // [1024,1024]
  const float* bvec   = (const float*)d_in[4];  // [1024]
  const float* btvec  = (const float*)d_in[5];  // [1024]
  const float* Wt     = (const float*)d_in[6];  // [1024,1024]
  const float* topics = (const float*)d_in[7];  // [32,1024]
  float* out = (float*)d_out;                   // [64,1024]

  // ws layout: bf16 region first, f32 scratch after
  __bf16* WTb    = (__bf16*)d_ws;               // [1024 x 1024] bf16 (transposed W)
  __bf16* WcTb   = WTb  + (size_t)DD * DD;
  __bf16* WtTb   = WcTb + (size_t)DD * DD;
  __bf16* cbf    = WtTb + (size_t)DD * DD;      // [64 x 1024] bf16
  __bf16* uitcbf = cbf  + (size_t)BB * DD;      // [64 x 1024] bf16
  float*  fws    = (float*)(uitcbf + (size_t)BB * DD);
  float* uitc = fws;                            // 64*1024 f32
  float* tij  = fws + BB * DD;                  // 64*1024 f32
  float* uw   = fws + 2 * BB * DD;              // 64*1024 f32
  float* e    = fws + 3 * BB * DD;              // 64*512  f32

  dim3 tg(32, 32);
  // one-time transpose+convert of the three weight matrices (L2-resident after)
  wprep_kernel<<<tg, 256, 0, stream>>>(Wc, WcTb);
  wprep_kernel<<<tg, 256, 0, stream>>>(Wt, WtTb);
  wprep_kernel<<<tg, 256, 0, stream>>>(W,  WTb);
  // c -> bf16 row-major
  abf_kernel<<<(BB * DD) / 256, 256, 0, stream>>>(c, cbf, BB * DD);

  // uit_c = c @ Wc   (also emits bf16 copy for next GEMM's A operand)
  gemm64_kernel<<<DD / GK_TN, 256, 0, stream>>>(cbf, WcTb, (const float*)nullptr,
                                                uitc, uitcbf, 0);
  // tij = tanh(uit_c @ Wt + bt)
  gemm64_kernel<<<DD / GK_TN, 256, 0, stream>>>(uitcbf, WtTb, btvec, tij,
                                                (__bf16*)nullptr, 1);
  // topic softmax -> uw
  topic_kernel<<<BB, 256, 0, stream>>>(tij, topics, uw);
  // fused x@W + tanh + dot(uw) -> e[b,t]
  attn_logits_kernel<<<dim3(TT / XC_TM, BB), 256, 0, stream>>>(x, WTb, bvec, uitc, uw, e);
  // softmax over T + a @ x -> out
  attn_out_kernel<<<BB, 256, 0, stream>>>(x, e, out);
}
